// AMMBottleneck_62904091018018
// MI455X (gfx1250) — matrix operand order repaired and checked by reading, not compiled, not run on hardware
//
#include <hip/hip_runtime.h>
#include <hip/hip_bf16.h>
#include <math.h>

typedef __attribute__((ext_vector_type(16))) _Float16 v16h;
typedef __attribute__((ext_vector_type(8)))  _Float16 v8h;
typedef __attribute__((ext_vector_type(8)))  float    v8f;

#define BATCH   32
#define C_IN    1024
#define WIDTH   256
#define OUT_C   1024
#define HW      196            // 14*14
#define BL      (BATCH*HW)     // 6272
#define KCW     16             // codewords per codebook
#define KD12    4096           // ncb*K for layers 1,2 (256*16)
#define KD3     1024           // ncb*K for layer 3   (64*16)
#define EPSBN   1e-5f

// ---------------------------------------------------------------------------
// lut[c][k][o] (f32) -> lutT[o][c*K+k] (f16), N-major so WMMA B-fragments are
// contiguous per lane.
// ---------------------------------------------------------------------------
__global__ void lut_cvt_kernel(const float* __restrict__ lut,
                               _Float16* __restrict__ lutT, int Kd, int Out) {
    int i = blockIdx.x * blockDim.x + threadIdx.x;
    if (i >= Kd * Out) return;
    int o  = i / Kd;
    int ck = i - o * Kd;
    lutT[(size_t)o * Kd + ck] = (_Float16)lut[(size_t)ck * Out + o];
}

// ---------------------------------------------------------------------------
// Layer-1 soft assignment: x[b, c*4+s, l] -> attn[(b*196+l)][c*16+k] (f16)
// ---------------------------------------------------------------------------
__global__ void attn1_kernel(const float* __restrict__ x,
                             const float* __restrict__ cen,
                             const float* __restrict__ invt,
                             _Float16* __restrict__ attn) {
    int b = blockIdx.x >> 8;          // 256 codebooks
    int c = blockIdx.x & 255;
    int l = threadIdx.x;
    if (l >= HW) return;
    const float* xb = x + ((size_t)b * C_IN + c * 4) * HW + l;
    float v0 = xb[0 * HW], v1 = xb[1 * HW], v2 = xb[2 * HW], v3 = xb[3 * HW];
    float it = invt[0];
    float sc[KCW], mx = -3.4e38f;
#pragma unroll
    for (int k = 0; k < KCW; ++k) {
        const float* ck = cen + ((size_t)c * KCW + k) * 4;
        float c0 = ck[0], c1 = ck[1], c2 = ck[2], c3 = ck[3];
        float dot = v0 * c0 + v1 * c1 + v2 * c2 + v3 * c3;
        float nsq = c0 * c0 + c1 * c1 + c2 * c2 + c3 * c3;
        float s = (2.0f * dot - nsq) * it;
        sc[k] = s;
        mx = fmaxf(mx, s);
    }
    float sum = 0.f;
#pragma unroll
    for (int k = 0; k < KCW; ++k) { sc[k] = __expf(sc[k] - mx); sum += sc[k]; }
    float inv = 1.0f / sum;
    v16h o;
#pragma unroll
    for (int k = 0; k < KCW; ++k) o[k] = (_Float16)(sc[k] * inv);
    *(v16h*)(attn + ((size_t)(b * HW + l)) * KD12 + c * KCW) = o;
}

// ---------------------------------------------------------------------------
// Layer-2: fused BN1+ReLU + 3x3 patch gather (pad=1) + soft assignment (sub=9)
// out1 is channel-major [B, 256, 196].
// ---------------------------------------------------------------------------
__global__ void attn2_kernel(const float* __restrict__ out1,
                             const float* __restrict__ cen,
                             const float* __restrict__ invt,
                             const float* __restrict__ g,
                             const float* __restrict__ bta,
                             const float* __restrict__ mean,
                             const float* __restrict__ istd,
                             _Float16* __restrict__ attn) {
    int b = blockIdx.x >> 8;
    int c = blockIdx.x & 255;
    int l = threadIdx.x;
    if (l >= HW) return;
    int h = l / 14, w = l - h * 14;
    float gm = g[c] * istd[c];
    float bb = bta[c] - mean[c] * gm;
    const float* base = out1 + ((size_t)b * WIDTH + c) * HW;
    float v[9];
#pragma unroll
    for (int i = 0; i < 3; ++i)
#pragma unroll
        for (int j = 0; j < 3; ++j) {
            int hh = h + i - 1, ww = w + j - 1;
            float val = 0.f;
            if (hh >= 0 && hh < 14 && ww >= 0 && ww < 14)
                val = fmaxf(base[hh * 14 + ww] * gm + bb, 0.f);
            v[i * 3 + j] = val;
        }
    float it = invt[0];
    float sc[KCW], mx = -3.4e38f;
#pragma unroll
    for (int k = 0; k < KCW; ++k) {
        const float* ck = cen + ((size_t)c * KCW + k) * 9;
        float dot = 0.f, nsq = 0.f;
#pragma unroll
        for (int s = 0; s < 9; ++s) { float cv = ck[s]; dot += v[s] * cv; nsq += cv * cv; }
        float s = (2.0f * dot - nsq) * it;
        sc[k] = s;
        mx = fmaxf(mx, s);
    }
    float sum = 0.f;
#pragma unroll
    for (int k = 0; k < KCW; ++k) { sc[k] = __expf(sc[k] - mx); sum += sc[k]; }
    float inv = 1.0f / sum;
    v16h o;
#pragma unroll
    for (int k = 0; k < KCW; ++k) o[k] = (_Float16)(sc[k] * inv);
    *(v16h*)(attn + ((size_t)(b * HW + l)) * KD12 + c * KCW) = o;
}

// ---------------------------------------------------------------------------
// Layer-3: fused BN2+ReLU + soft assignment (64 codebooks, sub=4)
// ---------------------------------------------------------------------------
__global__ void attn3_kernel(const float* __restrict__ out2,
                             const float* __restrict__ cen,
                             const float* __restrict__ invt,
                             const float* __restrict__ g,
                             const float* __restrict__ bta,
                             const float* __restrict__ mean,
                             const float* __restrict__ istd,
                             _Float16* __restrict__ attn) {
    int b = blockIdx.x >> 6;          // 64 codebooks
    int c = blockIdx.x & 63;
    int l = threadIdx.x;
    if (l >= HW) return;
    float v[4];
#pragma unroll
    for (int s = 0; s < 4; ++s) {
        int ch = c * 4 + s;
        float gm = g[ch] * istd[ch];
        float bb = bta[ch] - mean[ch] * gm;
        float raw = out2[((size_t)b * WIDTH + ch) * HW + l];
        v[s] = fmaxf(raw * gm + bb, 0.f);
    }
    float it = invt[0];
    float sc[KCW], mx = -3.4e38f;
#pragma unroll
    for (int k = 0; k < KCW; ++k) {
        const float* ck = cen + ((size_t)c * KCW + k) * 4;
        float dot = 0.f, nsq = 0.f;
#pragma unroll
        for (int s = 0; s < 4; ++s) { float cv = ck[s]; dot += v[s] * cv; nsq += cv * cv; }
        float s = (2.0f * dot - nsq) * it;
        sc[k] = s;
        mx = fmaxf(mx, s);
    }
    float sum = 0.f;
#pragma unroll
    for (int k = 0; k < KCW; ++k) { sc[k] = __expf(sc[k] - mx); sum += sc[k]; }
    float inv = 1.0f / sum;
    v16h o;
#pragma unroll
    for (int k = 0; k < KCW; ++k) o[k] = (_Float16)(sc[k] * inv);
    *(v16h*)(attn + ((size_t)(b * HW + l)) * KD3 + c * KCW) = o;
}

// ---------------------------------------------------------------------------
// Register-blocked WMMA GEMM: C[M=6272, N] = A[M,Kd](f16) x Bt[N,Kd](f16).
// One wave computes a 32x64 macro-tile = 2 M-tiles x 4 N-tiles (8 WMMAs per
// 32-deep k-step from 2 A-frags + 4 B-frags = 0.75 KB/wmma of L2 traffic).
// WMMAs issued serpentine so consecutive ops share one operand; the producer
// of each shared operand sets the CDNA5 reuse_a/reuse_b hint (OPSEL RA/RB).
// A-frag, contiguous k like B: lane(m=lane&15, half) = A[m][kk+half*16..+7] ++
// A[m][kk+half*16+8..+7]; B-frag: lane(n,half) = Bt[n][kk+half*16..+15].
// C scattered channel-major to out[b][n][l]  (m = b*196+l).
// ---------------------------------------------------------------------------
__global__ void gemm_wmma_kernel(const _Float16* __restrict__ A,
                                 const _Float16* __restrict__ Bt,
                                 float* __restrict__ C,
                                 int Kd, int Nch, int Ngroups, int totWaves) {
    int wave = threadIdx.x >> 5;
    int lane = threadIdx.x & 31;
    int w = blockIdx.x * (blockDim.x >> 5) + wave;
    if (w >= totWaves) return;
    int mg = w / Ngroups;               // 32-row group
    int ng = w - mg * Ngroups;          // 64-col group
    int half = lane >> 4;
    int lm = lane & 15;

    const _Float16* arow0 = A + (size_t)(mg * 32 + lm) * Kd;
    const _Float16* arow1 = arow0 + (size_t)16 * Kd;
    const _Float16* brow0 = Bt + (size_t)(ng * 64 + lm) * Kd;
    const _Float16* brow1 = brow0 + (size_t)16 * Kd;
    const _Float16* brow2 = brow0 + (size_t)32 * Kd;
    const _Float16* brow3 = brow0 + (size_t)48 * Kd;

    v8f a00 = {}, a01 = {}, a02 = {}, a03 = {};
    v8f a10 = {}, a11 = {}, a12 = {}, a13 = {};

    int aoff0 = half * 16;              // first 8-f16 chunk within k-step
    int aoff1 = half * 16 + 8;          // second chunk (contiguous with first)
    int boff  = half * 16;              // contiguous 16-f16 chunk

    for (int kk = 0; kk < Kd; kk += 32) {
        v8h l0, l1;
        l0 = *(const v8h*)(arow0 + kk + aoff0);
        l1 = *(const v8h*)(arow0 + kk + aoff1);
        v16h fa0 = __builtin_shufflevector(l0, l1,
                     0, 1, 2, 3, 4, 5, 6, 7, 8, 9, 10, 11, 12, 13, 14, 15);
        l0 = *(const v8h*)(arow1 + kk + aoff0);
        l1 = *(const v8h*)(arow1 + kk + aoff1);
        v16h fa1 = __builtin_shufflevector(l0, l1,
                     0, 1, 2, 3, 4, 5, 6, 7, 8, 9, 10, 11, 12, 13, 14, 15);
        v16h fb0 = *(const v16h*)(brow0 + kk + boff);
        v16h fb1 = *(const v16h*)(brow1 + kk + boff);
        v16h fb2 = *(const v16h*)(brow2 + kk + boff);
        v16h fb3 = *(const v16h*)(brow3 + kk + boff);

        // serpentine: each consecutive WMMA shares one operand with the last
        a00 = __builtin_amdgcn_wmma_f32_16x16x32_f16(false, fa0, false, fb0, (short)0, a00, false, true);
        a10 = __builtin_amdgcn_wmma_f32_16x16x32_f16(false, fa1, false, fb0, (short)0, a10, true,  false);
        a11 = __builtin_amdgcn_wmma_f32_16x16x32_f16(false, fa1, false, fb1, (short)0, a11, false, true);
        a01 = __builtin_amdgcn_wmma_f32_16x16x32_f16(false, fa0, false, fb1, (short)0, a01, true,  false);
        a02 = __builtin_amdgcn_wmma_f32_16x16x32_f16(false, fa0, false, fb2, (short)0, a02, false, true);
        a12 = __builtin_amdgcn_wmma_f32_16x16x32_f16(false, fa1, false, fb2, (short)0, a12, true,  false);
        a13 = __builtin_amdgcn_wmma_f32_16x16x32_f16(false, fa1, false, fb3, (short)0, a13, false, true);
        a03 = __builtin_amdgcn_wmma_f32_16x16x32_f16(false, fa0, false, fb3, (short)0, a03, false, false);
    }

    // epilogue: C/D layout -> lane(n=lm), VGPR r -> M = half*8 + r
    int mBase = mg * 32;
    int nBase = ng * 64 + lm;
    auto store_tile = [&](const v8f& acc, int mt, int nt) {
#pragma unroll
        for (int r = 0; r < 8; ++r) {
            int mm = mBase + mt * 16 + half * 8 + r;
            int bi = mm / HW;
            int l  = mm - bi * HW;
            C[((size_t)bi * Nch + nBase + nt * 16) * HW + l] = acc[r];
        }
    };
    store_tile(a00, 0, 0); store_tile(a01, 0, 1);
    store_tile(a02, 0, 2); store_tile(a03, 0, 3);
    store_tile(a10, 1, 0); store_tile(a11, 1, 1);
    store_tile(a12, 1, 2); store_tile(a13, 1, 3);
}

// ---------------------------------------------------------------------------
// Per-channel batch-norm statistics over (B, H, W): mean + rsqrt(var+eps)
// ---------------------------------------------------------------------------
__global__ void bnstats_kernel(const float* __restrict__ X, int Nch,
                               float* __restrict__ mean, float* __restrict__ istd) {
    int ch = blockIdx.x;
    __shared__ float ss[256];
    __shared__ float ss2[256];
    float s = 0.f, s2 = 0.f;
    for (int t = threadIdx.x; t < BL; t += blockDim.x) {
        int b = t / HW, l = t - b * HW;
        float v = X[((size_t)b * Nch + ch) * HW + l];
        s += v;
        s2 += v * v;
    }
    ss[threadIdx.x] = s;
    ss2[threadIdx.x] = s2;
    __syncthreads();
    for (int o = 128; o > 0; o >>= 1) {
        if ((int)threadIdx.x < o) {
            ss[threadIdx.x]  += ss[threadIdx.x + o];
            ss2[threadIdx.x] += ss2[threadIdx.x + o];
        }
        __syncthreads();
    }
    if (threadIdx.x == 0) {
        float m = ss[0] / (float)BL;
        float var = ss2[0] / (float)BL - m * m;
        mean[ch] = m;
        istd[ch] = rsqrtf(var + EPSBN);
    }
}

// ---------------------------------------------------------------------------
// Final: relu( bn3(out3) + identity )
// ---------------------------------------------------------------------------
__global__ void final_kernel(const float* __restrict__ out3,
                             const float* __restrict__ x,
                             const float* __restrict__ g,
                             const float* __restrict__ bta,
                             const float* __restrict__ mean,
                             const float* __restrict__ istd,
                             float* __restrict__ out) {
    size_t i = (size_t)blockIdx.x * blockDim.x + threadIdx.x;
    if (i >= (size_t)BATCH * OUT_C * HW) return;
    int ch = (int)((i / HW) & (OUT_C - 1));
    float gm = g[ch] * istd[ch];
    float v = out3[i] * gm + (bta[ch] - mean[ch] * gm) + x[i];
    out[i] = fmaxf(v, 0.f);
}

// ---------------------------------------------------------------------------
extern "C" void kernel_launch(void* const* d_in, const int* in_sizes, int n_in,
                              void* d_out, int out_size, void* d_ws, size_t ws_size,
                              hipStream_t stream) {
    const float* x       = (const float*)d_in[0];
    const float* c1_cen  = (const float*)d_in[1];
    const float* c1_lut  = (const float*)d_in[2];
    const float* c1_invt = (const float*)d_in[3];
    const float* c2_cen  = (const float*)d_in[4];
    const float* c2_lut  = (const float*)d_in[5];
    const float* c2_invt = (const float*)d_in[6];
    const float* c3_cen  = (const float*)d_in[7];
    const float* c3_lut  = (const float*)d_in[8];
    const float* c3_invt = (const float*)d_in[9];
    const float* bn1_g = (const float*)d_in[10];
    const float* bn1_b = (const float*)d_in[11];
    const float* bn2_g = (const float*)d_in[12];
    const float* bn2_b = (const float*)d_in[13];
    const float* bn3_g = (const float*)d_in[14];
    const float* bn3_b = (const float*)d_in[15];
    float* out = (float*)d_out;

    // workspace carve-up (stream-ordered reuse of attn & lut buffers)
    char* ws = (char*)d_ws;
    size_t off = 0;
    auto carve = [&](size_t bytes) {
        void* p = ws + off;
        off = (off + bytes + 255) & ~(size_t)255;
        return p;
    };
    _Float16* attn = (_Float16*)carve((size_t)BL * KD12 * 2);     // 51.4 MB (reused x3)
    _Float16* lutT = (_Float16*)carve((size_t)KD12 * WIDTH * 2);  // 2 MB   (reused x3)
    float* out1 = (float*)carve((size_t)BL * WIDTH * 4);          // 6.4 MB
    float* out2 = (float*)carve((size_t)BL * WIDTH * 4);          // 6.4 MB
    float* out3 = (float*)carve((size_t)BL * OUT_C * 4);          // 25.7 MB
    float* mean1 = (float*)carve(WIDTH * 4);
    float* istd1 = (float*)carve(WIDTH * 4);
    float* mean2 = (float*)carve(WIDTH * 4);
    float* istd2 = (float*)carve(WIDTH * 4);
    float* mean3 = (float*)carve(OUT_C * 4);
    float* istd3 = (float*)carve(OUT_C * 4);

    const int TPB = 256;
    const int ATTN_TPB = 224;            // 7 waves, covers 196 pixels
    const int GEMM_TPB = 128;            // 4 waves/block -> more blocks in flight

    // ---- layer 1: 1x1 AMM -> out1 [B,256,196]
    lut_cvt_kernel<<<(KD12 * WIDTH + TPB - 1) / TPB, TPB, 0, stream>>>(c1_lut, lutT, KD12, WIDTH);
    attn1_kernel<<<BATCH * 256, ATTN_TPB, 0, stream>>>(x, c1_cen, c1_invt, attn);
    {
        int ngr = WIDTH / 64;                       // 4
        int waves = (BL / 32) * ngr;                // 196*4 = 784
        gemm_wmma_kernel<<<(waves + 3) / 4, GEMM_TPB, 0, stream>>>(attn, lutT, out1,
                                                                   KD12, WIDTH, ngr, waves);
    }
    bnstats_kernel<<<WIDTH, TPB, 0, stream>>>(out1, WIDTH, mean1, istd1);

    // ---- layer 2: 3x3 AMM (BN1+ReLU+patches fused into attn2) -> out2
    attn2_kernel<<<BATCH * 256, ATTN_TPB, 0, stream>>>(out1, c2_cen, c2_invt,
                                                       bn1_g, bn1_b, mean1, istd1, attn);
    lut_cvt_kernel<<<(KD12 * WIDTH + TPB - 1) / TPB, TPB, 0, stream>>>(c2_lut, lutT, KD12, WIDTH);
    {
        int ngr = WIDTH / 64;
        int waves = (BL / 32) * ngr;
        gemm_wmma_kernel<<<(waves + 3) / 4, GEMM_TPB, 0, stream>>>(attn, lutT, out2,
                                                                   KD12, WIDTH, ngr, waves);
    }
    bnstats_kernel<<<WIDTH, TPB, 0, stream>>>(out2, WIDTH, mean2, istd2);

    // ---- layer 3: 1x1 AMM (BN2+ReLU fused into attn3) -> out3 [B,1024,196]
    attn3_kernel<<<BATCH * 64, ATTN_TPB, 0, stream>>>(out2, c3_cen, c3_invt,
                                                      bn2_g, bn2_b, mean2, istd2, attn);
    lut_cvt_kernel<<<(KD3 * OUT_C + TPB - 1) / TPB, TPB, 0, stream>>>(c3_lut, lutT, KD3, OUT_C);
    {
        int ngr = OUT_C / 64;                       // 16
        int waves = (BL / 32) * ngr;                // 196*16 = 3136
        gemm_wmma_kernel<<<(waves + 3) / 4, GEMM_TPB, 0, stream>>>(attn, lutT, out3,
                                                                   KD3, OUT_C, ngr, waves);
    }
    bnstats_kernel<<<OUT_C, TPB, 0, stream>>>(out3, OUT_C, mean3, istd3);

    // ---- bn3 + residual + relu
    {
        size_t n = (size_t)BATCH * OUT_C * HW;
        final_kernel<<<(unsigned)((n + TPB - 1) / TPB), TPB, 0, stream>>>(out3, x, bn3_g, bn3_b,
                                                                          mean3, istd3, out);
    }
}
